// EncoderDecoder_73023033966903
// MI455X (gfx1250) — matrix operand order repaired and checked by reading, not compiled, not run on hardware
//
#include <hip/hip_runtime.h>
#include <hip/hip_bf16.h>

// ---------------- problem constants ----------------
#define BB 256   // batch
#define SS 256   // encoder steps
#define TT 32    // decoder steps
#define HH 512   // hidden
#define EE 128   // embed
#define VV 1000  // vocab
#define AA 8     // actions
#define NTT 80   // targets
#define KTOT 640        // E + H (fused LSTM input)
#define ZN 2048         // 4*H gate columns

typedef __attribute__((ext_vector_type(8)))  float   v8f;
typedef __attribute__((ext_vector_type(16))) __bf16  v16bf;

union ABu { v16bf v; uint4 q[2]; };

__device__ __forceinline__ unsigned short f2bf(float f) {
    unsigned u = __float_as_uint(f);
    unsigned r = u + 0x7FFFu + ((u >> 16) & 1u);   // round-to-nearest-even
    return (unsigned short)(r >> 16);
}
__device__ __forceinline__ float bf2f(unsigned short h) {
    return __uint_as_float(((unsigned)h) << 16);
}
__device__ __forceinline__ float sigm(float x) { return 1.0f / (1.0f + __expf(-x)); }
__device__ __forceinline__ uint4 ld16(const unsigned short* p) { return *(const uint4*)p; }

// ---------------- prep: pack [Wi_e | Wh_e]^T into bf16 Wt[KTOT][ZN] ----------------
__global__ void k_prep_w(const float* __restrict__ Wi, const float* __restrict__ Wh,
                         unsigned short* __restrict__ Wt) {
    int t = blockIdx.x * blockDim.x + threadIdx.x;        // over KTOT*ZN
    if (t >= KTOT * ZN) return;
    int k = t / ZN, n = t % ZN;
    float v = (k < EE) ? Wi[(size_t)n * EE + k] : Wh[(size_t)n * HH + (k - EE)];
    Wt[t] = f2bf(v);
}

// ---------------- prep: embedding gather -> X[s][b][e] (bf16) ----------------
__global__ void k_embed(const int* __restrict__ enc_in, const float* __restrict__ emb,
                        unsigned short* __restrict__ X) {
    int t = blockIdx.x * blockDim.x + threadIdx.x;        // over SS*BB*EE
    int e = t & (EE - 1);
    int sb = t / EE;
    int b = sb & (BB - 1);
    int s = sb / BB;
    int idx = enc_in[b * SS + s];                         // encoder_input is (B,S)
    X[t] = f2bf(emb[(size_t)idx * EE + e]);
}

// ---------------- prep: zero c (f32) and h0 (bf16) ----------------
__global__ void k_zero(float* __restrict__ c, unsigned short* __restrict__ h0) {
    int t = blockIdx.x * blockDim.x + threadIdx.x;        // over BB*HH
    c[t] = 0.0f;
    h0[t] = 0;
}

// ---------------- encoder LSTM step: WMMA GEMM + gates ----------------
// grid 256 = 16 m-tiles x 16 hidden-column groups; 256 threads = 8 waves.
// wave w: gate = w>>1, sub = w&1 -> z tile at columns gate*512 + hg*32 + sub*16.
// K-loop split into affine X-phase (K=0..127) and H-phase (K=128..639) so the
// compiler emits pure offset-bump addressing instead of per-iter selects.
__global__ __launch_bounds__(256) void k_enc_step(
    const unsigned short* __restrict__ Xs,     // [B][E]  bf16 (this step's input)
    const unsigned short* __restrict__ Hprev,  // [B][H]  bf16 (previous h)
    const unsigned short* __restrict__ Wt,     // [KTOT][ZN] bf16
    const float* __restrict__ be,              // [ZN] bias
    float* __restrict__ c,                     // [B][H] f32 cell state (in place)
    unsigned short* __restrict__ Hout)         // [B][H] bf16 (= enc_out[s])
{
    __shared__ float zbuf[4][16][33];          // +1 pad vs LDS bank conflicts

    const int lane = threadIdx.x & 31;
    const int wv   = threadIdx.x >> 5;
    const int gate = wv >> 1;
    const int sub  = wv & 1;
    const int mt   = blockIdx.x & 15;
    const int hg   = blockIdx.x >> 4;
    const int n0   = gate * HH + hg * 32 + sub * 16;

    const int mrow = mt * 16 + (lane & 15);    // A-matrix row = batch index
    const int hi   = lane >> 4;                // ISA 16-bit A layout half-select

    v8f acc = {};

    // ---- phase 1: K = 0..127 from X_s ----
    {
        const unsigned short* arow = Xs + (size_t)mrow * EE + hi * 8;
        const unsigned short* bp   = Wt + (size_t)(hi * 8) * ZN + n0 + (lane & 15);
        #pragma unroll
        for (int kc = 0; kc < EE / 32; ++kc) {
            ABu aop, bop;
            aop.q[0] = ld16(arow + kc * 32);
            aop.q[1] = ld16(arow + kc * 32 + 16);
            #pragma unroll
            for (int s = 0; s < 8; ++s) {
                ((unsigned short*)&bop)[s]     = bp[(size_t)(kc * 32 + s) * ZN];
                ((unsigned short*)&bop)[8 + s] = bp[(size_t)(kc * 32 + 16 + s) * ZN];
            }
            acc = __builtin_amdgcn_wmma_f32_16x16x32_bf16(
                false, aop.v, false, bop.v, (short)0, acc, false, false);
        }
    }

    // ---- phase 2: K = 128..639 from h_{s-1} ----
    {
        const unsigned short* arow = Hprev + (size_t)mrow * HH + hi * 8;
        const unsigned short* bp   = Wt + ((size_t)EE + hi * 8) * ZN + n0 + (lane & 15);
        #pragma unroll 4
        for (int kc = 0; kc < HH / 32; ++kc) {
            ABu aop, bop;
            aop.q[0] = ld16(arow + kc * 32);
            aop.q[1] = ld16(arow + kc * 32 + 16);
            #pragma unroll
            for (int s = 0; s < 8; ++s) {
                ((unsigned short*)&bop)[s]     = bp[(size_t)(kc * 32 + s) * ZN];
                ((unsigned short*)&bop)[8 + s] = bp[(size_t)(kc * 32 + 16 + s) * ZN];
            }
            if (kc + 1 < HH / 32)   // keep next K rows of B warm in near caches
                __builtin_prefetch(bp + (size_t)(kc + 1) * 32 * ZN, 0, 3);
            acc = __builtin_amdgcn_wmma_f32_16x16x32_bf16(
                false, aop.v, false, bop.v, (short)0, acc, false, false);
        }
    }

    // scatter 16x16 f32 tile (ISA C/D layout) into LDS
    const int col = sub * 16 + (lane & 15);
    #pragma unroll
    for (int r = 0; r < 8; ++r)
        zbuf[gate][hi * 8 + r][col] = acc[r];
    __syncthreads();

    // gate math: 16 rows x 32 cols = 512 elems, 2 per thread
    #pragma unroll
    for (int rep = 0; rep < 2; ++rep) {
        int e  = threadIdx.x + rep * 256;
        int lr = e >> 5, lc = e & 31;
        int bi = mt * 16 + lr;
        int j  = hg * 32 + lc;
        float iv = zbuf[0][lr][lc] + be[j];
        float fv = zbuf[1][lr][lc] + be[HH + j];
        float gv = zbuf[2][lr][lc] + be[2 * HH + j];
        float ov = zbuf[3][lr][lc] + be[3 * HH + j];
        float co = c[(size_t)bi * HH + j];
        float cn = sigm(fv) * co + sigm(iv) * tanhf(gv);
        float hn = sigm(ov) * tanhf(cn);
        c[(size_t)bi * HH + j] = cn;
        Hout[(size_t)bi * HH + j] = f2bf(hn);
    }
}

// ---------------- attention: enc_score + softmax over S (per batch column) ----------------
// NOTE: softmax(enc_score + h@Wd + bf, axis=S) == softmax(enc_score, axis=S) exactly,
// since the decoder term is constant along S. The decoder LSTM never affects outputs.
__global__ __launch_bounds__(256) void k_score_softmax(
    const unsigned short* __restrict__ enc_out,  // [S][B][H] bf16
    const float* __restrict__ Wf,                // (1,2H): first H = We
    float* __restrict__ w)                       // [S][B]
{
    int b = blockIdx.x, s = threadIdx.x;
    const unsigned short* row = enc_out + ((size_t)s * BB + b) * HH;
    float acc = 0.0f;
    for (int h = 0; h < HH; ++h) acc += bf2f(row[h]) * Wf[h];

    __shared__ float sm[256];
    sm[s] = acc; __syncthreads();
    for (int off = 128; off > 0; off >>= 1) {
        if (s < off) sm[s] = fmaxf(sm[s], sm[s + off]);
        __syncthreads();
    }
    float mx = sm[0]; __syncthreads();
    float ex = __expf(acc - mx);
    sm[s] = ex; __syncthreads();
    for (int off = 128; off > 0; off >>= 1) {
        if (s < off) sm[s] += sm[s + off];
        __syncthreads();
    }
    w[s * BB + b] = ex / sm[0];
}

// ---------------- context: ctx[b][h] = sum_s w[s][b] * enc_out[s][b][h] ----------------
__global__ __launch_bounds__(256) void k_ctx(
    const unsigned short* __restrict__ enc_out, const float* __restrict__ w,
    float* __restrict__ ctx)
{
    int b = blockIdx.x, h = threadIdx.x;
    float a0 = 0.0f, a1 = 0.0f;
    for (int s = 0; s < SS; ++s) {
        float ws = w[s * BB + b];
        const unsigned short* row = enc_out + ((size_t)s * BB + b) * HH;
        a0 += ws * bf2f(row[h]);
        a1 += ws * bf2f(row[h + 256]);
    }
    ctx[(size_t)b * HH + h] = a0;
    ctx[(size_t)b * HH + h + 256] = a1;
}

// ---------------- heads: pa/pt from ctx, broadcast over all T steps ----------------
__global__ __launch_bounds__(128) void k_heads(
    const float* __restrict__ ctx,
    const float* __restrict__ Wa, const float* __restrict__ ba,
    const float* __restrict__ Wt2, const float* __restrict__ bt,
    float* __restrict__ out)
{
    int b = blockIdx.x;
    __shared__ float cx[HH];
    for (int i = threadIdx.x; i < HH; i += 128) cx[i] = ctx[(size_t)b * HH + i];
    __syncthreads();
    int o = threadIdx.x;
    if (o < AA) {
        float acc = ba[o];
        for (int h = 0; h < HH; ++h) acc += cx[h] * Wa[(size_t)o * HH + h];
        float* pa = out + (size_t)b * TT * AA;
        for (int t = 0; t < TT; ++t) pa[t * AA + o] = acc;
    } else if (o < AA + NTT) {
        int oo = o - AA;
        float acc = bt[oo];
        for (int h = 0; h < HH; ++h) acc += cx[h] * Wt2[(size_t)oo * HH + h];
        float* pt = out + (size_t)BB * TT * AA + (size_t)b * TT * NTT;
        for (int t = 0; t < TT; ++t) pt[t * NTT + oo] = acc;
    }
}

// ---------------- host launch ----------------
extern "C" void kernel_launch(void* const* d_in, const int* in_sizes, int n_in,
                              void* d_out, int out_size, void* d_ws, size_t ws_size,
                              hipStream_t stream) {
    const int*   enc_in = (const int*)  d_in[0];
    const float* emb    = (const float*)d_in[2];
    const float* Wi_e   = (const float*)d_in[3];
    const float* Wh_e   = (const float*)d_in[4];
    const float* b_e    = (const float*)d_in[5];
    const float* W_a    = (const float*)d_in[11];
    const float* b_a    = (const float*)d_in[12];
    const float* W_t    = (const float*)d_in[13];
    const float* b_t    = (const float*)d_in[14];
    const float* Wf     = (const float*)d_in[15];
    float* out = (float*)d_out;

    char* ws = (char*)d_ws;
    size_t off = 0;
    auto carve = [&](size_t bytes) -> void* {
        void* p = ws + off;
        off += (bytes + 255) & ~(size_t)255;
        return p;
    };
    unsigned short* Wt_enc  = (unsigned short*)carve((size_t)KTOT * ZN * 2);
    unsigned short* X       = (unsigned short*)carve((size_t)SS * BB * EE * 2);
    unsigned short* enc_out = (unsigned short*)carve((size_t)SS * BB * HH * 2);
    unsigned short* h0      = (unsigned short*)carve((size_t)BB * HH * 2);
    float*          cst     = (float*)         carve((size_t)BB * HH * 4);
    float*          wsm     = (float*)         carve((size_t)SS * BB * 4);
    float*          ctx     = (float*)         carve((size_t)BB * HH * 4);
    (void)ws_size; (void)in_sizes; (void)n_in; (void)out_size;

    k_prep_w<<<(KTOT * ZN) / 256, 256, 0, stream>>>(Wi_e, Wh_e, Wt_enc);
    k_embed <<<(SS * BB * EE) / 256, 256, 0, stream>>>(enc_in, emb, X);
    k_zero  <<<(BB * HH) / 256, 256, 0, stream>>>(cst, h0);

    for (int s = 0; s < SS; ++s) {
        const unsigned short* Hprev = (s == 0) ? h0 : enc_out + (size_t)(s - 1) * BB * HH;
        k_enc_step<<<256, 256, 0, stream>>>(
            X + (size_t)s * BB * EE, Hprev, Wt_enc, b_e, cst,
            enc_out + (size_t)s * BB * HH);
    }

    k_score_softmax<<<BB, 256, 0, stream>>>(enc_out, Wf, wsm);
    k_ctx          <<<BB, 256, 0, stream>>>(enc_out, wsm, ctx);
    k_heads        <<<BB, 128, 0, stream>>>(ctx, W_a, b_a, W_t, b_t, out);
}
